// Self_Attn_88227218194668
// MI455X (gfx1250) — compile-verified, matrix operand order checked by reading, not run on hardware
//
#include <hip/hip_runtime.h>
#include <hip/hip_bf16.h>

// ---------------------------------------------------------------------------
// Self-attention (pooled) for MI455X / gfx1250, wave32 + v_wmma_f32_16x16x32_bf16
// B=8, C=512, H=W=128 -> pooled 64x64 (N=4096), head dim 64, V dim 512.
// Fused flash-attention (online softmax) so the 4096x4096 attention matrix is
// never materialized; all GEMMs via WMMA bf16 with f32 accumulation.
// Fragment loads are software-pipelined (double-buffered) so WMMAs overlap
// with in-flight global loads instead of draining loadcnt to 0 each time.
// ---------------------------------------------------------------------------

constexpr int kB  = 8;
constexpr int kC  = 512;
constexpr int kCQ = 64;     // C/8
constexpr int kH  = 128;
constexpr int kW  = 128;
constexpr int kHp = 64;
constexpr int kN  = 4096;   // kHp * kWp

typedef __attribute__((ext_vector_type(16))) __bf16 bf16x16;
typedef __attribute__((ext_vector_type(8)))  __bf16 bf16x8;
typedef __attribute__((ext_vector_type(8)))  float  f32x8;

union ABFrag { bf16x16 v; bf16x8 h[2]; };

// Load a 16-element bf16 fragment as two 16-byte chunks (per-lane A/B frag).
__device__ __forceinline__ bf16x16 load_frag(const __bf16* p0, const __bf16* p1) {
  ABFrag f;
  f.h[0] = *(const bf16x8*)p0;
  f.h[1] = *(const bf16x8*)p1;
  return f.v;
}

__device__ __forceinline__ f32x8 wmma_bf16(bf16x16 a, bf16x16 b, f32x8 c) {
  return __builtin_amdgcn_wmma_f32_16x16x32_bf16(
      /*neg_a=*/false, a, /*neg_b=*/false, b,
      /*c_mod=*/(short)0, c, /*reuse_a=*/false, /*reuse_b=*/false);
}

// Reduce across the 16 lanes of a half-wave (lanes 0-15 or 16-31 independently).
__device__ __forceinline__ float half_reduce_max(float v) {
  v = fmaxf(v, __shfl_xor(v, 1, 32));
  v = fmaxf(v, __shfl_xor(v, 2, 32));
  v = fmaxf(v, __shfl_xor(v, 4, 32));
  v = fmaxf(v, __shfl_xor(v, 8, 32));
  return v;
}
__device__ __forceinline__ float half_reduce_sum(float v) {
  v += __shfl_xor(v, 1, 32);
  v += __shfl_xor(v, 2, 32);
  v += __shfl_xor(v, 4, 32);
  v += __shfl_xor(v, 8, 32);
  return v;
}

// ---------------------------------------------------------------------------
// K0: convert projection weights f32 -> bf16
// ---------------------------------------------------------------------------
__global__ __launch_bounds__(256)
void cvt_weights(const float* __restrict__ wq, const float* __restrict__ wk,
                 const float* __restrict__ wv,
                 __bf16* __restrict__ wqb, __bf16* __restrict__ wkb,
                 __bf16* __restrict__ wvb) {
  int i = blockIdx.x * 256 + threadIdx.x;
  if (i < kCQ * kC) {
    wqb[i] = (__bf16)wq[i];
    wkb[i] = (__bf16)wk[i];
  }
  if (i < kC * kC) wvb[i] = (__bf16)wv[i];
}

// ---------------------------------------------------------------------------
// K1: AvgPool2d(2) + LayerNorm over width(64) -> XN (B, N, C) bf16.
// Block handles one (b, h, 64-channel tile); LDS-transposed so both the
// f32 reads of x and the bf16 writes of XN are contiguous.
// ---------------------------------------------------------------------------
__global__ __launch_bounds__(256)
void pool_ln(const float* __restrict__ x, const float* __restrict__ lnw,
             const float* __restrict__ lnb, __bf16* __restrict__ XN) {
  const int bid = blockIdx.x;
  const int ct = bid & 7;
  const int h  = (bid >> 3) & 63;
  const int b  = bid >> 9;
  const int c0 = ct * 64;
  const int tid = threadIdx.x;

  __shared__ float tile[64][65];
  __shared__ float mu_s[64], rs_s[64];

  #pragma unroll
  for (int it = 0; it < 16; ++it) {
    int idx = it * 256 + tid;
    int cl = idx >> 6, w = idx & 63;
    const float* px = x + ((size_t)(b * kC + c0 + cl) * kH + 2 * h) * kW + 2 * w;
    float2 r0 = *(const float2*)px;
    float2 r1 = *(const float2*)(px + kW);
    tile[cl][w] = 0.25f * (r0.x + r0.y + r1.x + r1.y);
  }
  __syncthreads();
  if (tid < 64) {
    float s = 0.f, s2 = 0.f;
    for (int w = 0; w < 64; ++w) { float v = tile[tid][w]; s += v; s2 += v * v; }
    float mu  = s * 0.015625f;
    float var = s2 * 0.015625f - mu * mu;
    mu_s[tid] = mu;
    rs_s[tid] = rsqrtf(var + 1e-5f);
  }
  __syncthreads();
  #pragma unroll
  for (int it = 0; it < 16; ++it) {
    int idx = it * 256 + tid;
    int cl = idx & 63, w = idx >> 6;
    float v = (tile[cl][w] - mu_s[cl]) * rs_s[cl] * lnw[w] + lnb[w];
    XN[((size_t)b * kN + h * 64 + w) * kC + c0 + cl] = (__bf16)v;
  }
}

// ---------------------------------------------------------------------------
// K2: Q/K projections.  out[n,o] = sum_c XN[b,n,c] * Wq[o,c] + bq[o].
// A = XN rows (K contiguous), B columns = weight rows (K contiguous).
// Block = 4 waves; each wave: 16 n-rows x 64 outputs for both Q and K.
// B/A fragments are prefetched one step ahead of the consuming WMMA.
// ---------------------------------------------------------------------------
__global__ __launch_bounds__(128)
void proj_qk(const __bf16* __restrict__ XN, const __bf16* __restrict__ WQb,
             const __bf16* __restrict__ WKb, const float* __restrict__ bq,
             const float* __restrict__ bk, __bf16* __restrict__ Qw,
             __bf16* __restrict__ Kw) {
  const int b = blockIdx.y;
  const int tid = threadIdx.x, wave = tid >> 5, lane = tid & 31;
  const int col = lane & 15, kh = (lane >> 4) << 3;
  const int n0 = blockIdx.x * 64 + wave * 16;
  constexpr int kKS = kC / 32;

  f32x8 z = {0, 0, 0, 0, 0, 0, 0, 0};
  f32x8 qa[4], ka[4];
  #pragma unroll
  for (int t = 0; t < 4; ++t) { qa[t] = z; ka[t] = z; }

  const __bf16* abase = XN + ((size_t)b * kN + n0 + col) * kC;
  const __bf16* qbase = WQb + (size_t)col * kC;
  const __bf16* kbase = WKb + (size_t)col * kC;

  bf16x16 a   = load_frag(abase + kh, abase + kh + 16);
  bf16x16 bqf = load_frag(qbase + kh, qbase + kh + 16);
  bf16x16 bkf = load_frag(kbase + kh, kbase + kh + 16);

  for (int ks = 0; ks < kKS; ++ks) {
    const int ko = ks * 32;
    bf16x16 a_cur = a;
    if (ks + 1 < kKS) {
      const __bf16* ap = abase + ko + 32;
      a = load_frag(ap + kh, ap + kh + 16);
    }
    #pragma unroll
    for (int t = 0; t < 4; ++t) {
      bf16x16 q_cur = bqf, k_cur = bkf;
      if (t < 3) {
        const __bf16* nq = qbase + (size_t)(t + 1) * 16 * kC + ko;
        const __bf16* nk = kbase + (size_t)(t + 1) * 16 * kC + ko;
        bqf = load_frag(nq + kh, nq + kh + 16);
        bkf = load_frag(nk + kh, nk + kh + 16);
      } else if (ks + 1 < kKS) {
        const __bf16* nq = qbase + ko + 32;
        const __bf16* nk = kbase + ko + 32;
        bqf = load_frag(nq + kh, nq + kh + 16);
        bkf = load_frag(nk + kh, nk + kh + 16);
      }
      qa[t] = wmma_bf16(a_cur, q_cur, qa[t]);
      ka[t] = wmma_bf16(a_cur, k_cur, ka[t]);
    }
  }
  #pragma unroll
  for (int t = 0; t < 4; ++t) {
    int o = t * 16 + col;
    float bqv = bq[o], bkv = bk[o];
    #pragma unroll
    for (int j = 0; j < 8; ++j) {
      size_t row = (size_t)b * kN + n0 + j + kh;
      Qw[row * kCQ + o] = (__bf16)(qa[t][j] + bqv);
      Kw[row * kCQ + o] = (__bf16)(ka[t][j] + bkv);
    }
  }
}

// ---------------------------------------------------------------------------
// K3: V projection. v[b,c,n] = sum_c' Wv[c,c'] * XN[b,n,c'] + bv[c].
// A = Wv rows (K contiguous); B columns = XN rows (K contiguous).
// Output stored (B, C, N) bf16 so attention PV B-frags are m-contiguous.
// ---------------------------------------------------------------------------
__global__ __launch_bounds__(128)
void proj_v(const __bf16* __restrict__ XN, const __bf16* __restrict__ WVb,
            const float* __restrict__ bv, __bf16* __restrict__ Vw) {
  const int b = blockIdx.y >> 3, ct = blockIdx.y & 7;
  const int tid = threadIdx.x, wave = tid >> 5, lane = tid & 31;
  const int col = lane & 15, kh = (lane >> 4) << 3;
  const int c0 = ct * 64 + wave * 16;
  const int n0 = blockIdx.x * 64;
  constexpr int kKS = kC / 32;

  f32x8 z = {0, 0, 0, 0, 0, 0, 0, 0};
  f32x8 acc[4];
  #pragma unroll
  for (int t = 0; t < 4; ++t) acc[t] = z;

  const __bf16* abase = WVb + (size_t)(c0 + col) * kC;
  const __bf16* bbase = XN + ((size_t)b * kN + n0 + col) * kC;

  bf16x16 a  = load_frag(abase + kh, abase + kh + 16);
  bf16x16 bf = load_frag(bbase + kh, bbase + kh + 16);

  for (int ks = 0; ks < kKS; ++ks) {
    const int ko = ks * 32;
    bf16x16 a_cur = a;
    if (ks + 1 < kKS) {
      const __bf16* ap = abase + ko + 32;
      a = load_frag(ap + kh, ap + kh + 16);
    }
    #pragma unroll
    for (int t = 0; t < 4; ++t) {
      bf16x16 b_cur = bf;
      if (t < 3) {
        const __bf16* bp = bbase + (size_t)(t + 1) * 16 * kC + ko;
        bf = load_frag(bp + kh, bp + kh + 16);
      } else if (ks + 1 < kKS) {
        const __bf16* bp = bbase + ko + 32;
        bf = load_frag(bp + kh, bp + kh + 16);
      }
      acc[t] = wmma_bf16(a_cur, b_cur, acc[t]);
    }
  }
  #pragma unroll
  for (int j = 0; j < 8; ++j) {
    int c = c0 + j + kh;
    float bvv = bv[c];
    #pragma unroll
    for (int t = 0; t < 4; ++t) {
      Vw[((size_t)b * kC + c) * kN + n0 + t * 16 + col] = (__bf16)(acc[t][j] + bvv);
    }
  }
}

// ---------------------------------------------------------------------------
// K4: Flash attention.  Block = 4 waves, 16 queries per block.
// Per 64-key step: each wave computes one 16x16 energy tile (2 wmma),
// cross-wave online softmax through LDS, P staged in LDS (A-frag layout),
// each wave accumulates 16x128 of the 512 output channels (16 wmma).
// V fragments double-buffered across the 8 PV tiles; next-step K fragments
// prefetched during the PV tail so WMMAs never drain loadcnt to zero.
// ---------------------------------------------------------------------------
__global__ __launch_bounds__(128)
void flash_attn(const __bf16* __restrict__ Qw, const __bf16* __restrict__ Kw,
                const __bf16* __restrict__ Vw, float* __restrict__ Ow) {
  const int b  = blockIdx.y;
  const int q0 = blockIdx.x << 4;
  const int tid = threadIdx.x, wave = tid >> 5, lane = tid & 31;
  const int col = lane & 15, kh = (lane >> 4) << 3;  // kh = hi*8 (row offset too)

  __shared__ float pmax[4][16];
  __shared__ float psum[4][16];
  __shared__ __align__(16) __bf16 Plds[16 * 64];

  // Preload Q A-fragments for rows q0..q0+15, d = 0..63 (two 16x32 frags).
  const __bf16* qb = Qw + ((size_t)b * kN + q0 + col) * kCQ;
  bf16x16 qf0 = load_frag(qb + kh,      qb + kh + 16);
  bf16x16 qf1 = load_frag(qb + 32 + kh, qb + 48 + kh);

  f32x8 z = {0, 0, 0, 0, 0, 0, 0, 0};
  f32x8 oacc[8];
  #pragma unroll
  for (int t = 0; t < 8; ++t) oacc[t] = z;

  float rowM[8], rowL[8];
  #pragma unroll
  for (int j = 0; j < 8; ++j) { rowM[j] = -3.0e38f; rowL[j] = 0.f; }

  constexpr int kSteps = kN / 64;

  // Prefetch K fragments for step 0.
  const __bf16* kb0 = Kw + ((size_t)b * kN + wave * 16 + col) * kCQ;
  bf16x16 kf0 = load_frag(kb0 + kh,      kb0 + kh + 16);
  bf16x16 kf1 = load_frag(kb0 + 32 + kh, kb0 + 48 + kh);

  for (int kt = 0; kt < kSteps; ++kt) {
    const int m0 = kt * 64;

    // Energy tile: columns m0 + wave*16 + [0..15], K-dim = 64.
    f32x8 e = z;
    e = wmma_bf16(qf0, kf0, e);
    e = wmma_bf16(qf1, kf1, e);

    // Prefetch first V tile of this step (overlaps the softmax VALU work).
    const __bf16* vbase = Vw + ((size_t)b * kC + wave * 128 + col) * kN + m0;
    bf16x16 vc0 = load_frag(vbase + kh,      vbase + kh + 16);
    bf16x16 vc1 = load_frag(vbase + 32 + kh, vbase + 48 + kh);

    // Per-row partial max within this wave's 16 columns.
    float pm[8];
    #pragma unroll
    for (int j = 0; j < 8; ++j) pm[j] = half_reduce_max(e[j]);
    if (col == 0) {
      #pragma unroll
      for (int j = 0; j < 8; ++j) pmax[wave][j + kh] = pm[j];
    }
    __syncthreads();

    // Online softmax: new running max, rescale accumulators, exponentiate.
    float ps[8];
    #pragma unroll
    for (int j = 0; j < 8; ++j) {
      int r = j + kh;
      float g = fmaxf(fmaxf(pmax[0][r], pmax[1][r]), fmaxf(pmax[2][r], pmax[3][r]));
      float mNew = fmaxf(rowM[j], g);
      float cf = __expf(rowM[j] - mNew);
      rowM[j] = mNew;
      rowL[j] *= cf;
      #pragma unroll
      for (int t = 0; t < 8; ++t) oacc[t][j] *= cf;
      float pv = __expf(e[j] - mNew);
      Plds[r * 64 + wave * 16 + col] = (__bf16)pv;
      ps[j] = half_reduce_sum(pv);
    }
    if (col == 0) {
      #pragma unroll
      for (int j = 0; j < 8; ++j) psum[wave][j + kh] = ps[j];
    }
    __syncthreads();

    #pragma unroll
    for (int j = 0; j < 8; ++j) {
      int r = j + kh;
      rowL[j] += psum[0][r] + psum[1][r] + psum[2][r] + psum[3][r];
    }

    // P A-fragments (16 rows x 64 m) from LDS.
    const __bf16* pb = Plds + col * 64;
    bf16x16 pf0 = load_frag(pb + kh,      pb + kh + 16);
    bf16x16 pf1 = load_frag(pb + 32 + kh, pb + 48 + kh);

    // PV: this wave's 128 output channels (8 tiles of 16), K-dim = 64.
    // V fragments double-buffered: tile t+1 loads issue before tile t's WMMAs.
    #pragma unroll
    for (int t = 0; t < 8; ++t) {
      bf16x16 nx0 = vc0, nx1 = vc1;
      if (t < 7) {
        const __bf16* vp = vbase + (size_t)(t + 1) * 16 * kN;
        nx0 = load_frag(vp + kh,      vp + kh + 16);
        nx1 = load_frag(vp + 32 + kh, vp + 48 + kh);
      }
      oacc[t] = wmma_bf16(pf0, vc0, oacc[t]);
      oacc[t] = wmma_bf16(pf1, vc1, oacc[t]);
      vc0 = nx0; vc1 = nx1;
    }

    // Prefetch next step's K fragments (overlaps the barrier + next energy).
    if (kt + 1 < kSteps) {
      const __bf16* kb = Kw + ((size_t)b * kN + m0 + 64 + wave * 16 + col) * kCQ;
      kf0 = load_frag(kb + kh,      kb + kh + 16);
      kf1 = load_frag(kb + 32 + kh, kb + 48 + kh);
    }
    __syncthreads();  // protect Plds / psum before next iteration's writes
  }

  // Normalize by running sum and store O as (B, N, C) f32.
  #pragma unroll
  for (int j = 0; j < 8; ++j) {
    float inv = 1.0f / rowL[j];
    #pragma unroll
    for (int t = 0; t < 8; ++t) {
      Ow[((size_t)b * kN + q0 + j + kh) * kC + wave * 128 + t * 16 + col] =
          oacc[t][j] * inv;
    }
  }
}

// ---------------------------------------------------------------------------
// K5: LayerNorm(out) over width(64) + nearest 2x upsample + gamma*out + x.
// Block handles one (b, h); loops 8 channel tiles through an LDS transpose
// so both the O reads and the final f32 writes are contiguous.
// ---------------------------------------------------------------------------
__global__ __launch_bounds__(256)
void ln_out_up(const float* __restrict__ O, const float* __restrict__ x,
               const float* __restrict__ gamma, const float* __restrict__ lnw,
               const float* __restrict__ lnb, float* __restrict__ out) {
  const int b = blockIdx.x >> 6;
  const int h = blockIdx.x & 63;
  const int tid = threadIdx.x;
  const float g = gamma[0];

  __shared__ float tile[64][65];  // [w][c_local]
  __shared__ float mu_s[64], rs_s[64];

  for (int ct = 0; ct < 8; ++ct) {
    const int c0 = ct * 64;
    #pragma unroll
    for (int it = 0; it < 16; ++it) {
      int idx = it * 256 + tid;
      int w = idx >> 6, cl = idx & 63;
      tile[w][cl] = O[((size_t)b * kN + h * 64 + w) * kC + c0 + cl];
    }
    __syncthreads();
    if (tid < 64) {
      float s = 0.f, s2 = 0.f;
      for (int w = 0; w < 64; ++w) { float v = tile[w][tid]; s += v; s2 += v * v; }
      float mu  = s * 0.015625f;
      float var = s2 * 0.015625f - mu * mu;
      mu_s[tid] = mu;
      rs_s[tid] = rsqrtf(var + 1e-5f);
    }
    __syncthreads();
    #pragma unroll
    for (int it = 0; it < 16; ++it) {
      int idx = it * 256 + tid;
      int cl = idx >> 6, w = idx & 63;
      float v  = (tile[w][cl] - mu_s[cl]) * rs_s[cl] * lnw[w] + lnb[w];
      float gv = g * v;
      size_t base = ((size_t)(b * kC + c0 + cl) * kH + 2 * h) * kW + 2 * w;
      float2 x0 = *(const float2*)(x + base);
      float2 x1 = *(const float2*)(x + base + kW);
      *(float2*)(out + base)      = make_float2(gv + x0.x, gv + x0.y);
      *(float2*)(out + base + kW) = make_float2(gv + x1.x, gv + x1.y);
    }
    __syncthreads();
  }
}

// ---------------------------------------------------------------------------
// Host launcher
// ---------------------------------------------------------------------------
extern "C" void kernel_launch(void* const* d_in, const int* in_sizes, int n_in,
                              void* d_out, int out_size, void* d_ws, size_t ws_size,
                              hipStream_t stream) {
  (void)in_sizes; (void)n_in; (void)out_size; (void)ws_size;
  const float* x     = (const float*)d_in[0];
  const float* wq    = (const float*)d_in[1];
  const float* bq    = (const float*)d_in[2];
  const float* wk    = (const float*)d_in[3];
  const float* bk    = (const float*)d_in[4];
  const float* wv    = (const float*)d_in[5];
  const float* bv    = (const float*)d_in[6];
  const float* gamma = (const float*)d_in[7];
  const float* lniw  = (const float*)d_in[8];
  const float* lnib  = (const float*)d_in[9];
  const float* lnow  = (const float*)d_in[10];
  const float* lnob  = (const float*)d_in[11];
  float* out = (float*)d_out;

  char* ws = (char*)d_ws;
  size_t off = 0;
  auto alloc = [&](size_t bytes) -> void* {
    void* p = ws + off;
    off += (bytes + 255) & ~(size_t)255;
    return p;
  };
  __bf16* XN  = (__bf16*)alloc((size_t)kB * kN * kC * 2);   // pooled+LN, (B,N,C)
  __bf16* Qw  = (__bf16*)alloc((size_t)kB * kN * kCQ * 2);  // (B,N,64)
  __bf16* Kw  = (__bf16*)alloc((size_t)kB * kN * kCQ * 2);  // (B,N,64)
  __bf16* Vw  = (__bf16*)alloc((size_t)kB * kC * kN * 2);   // (B,C,N)
  float*  Ow  = (float*) alloc((size_t)kB * kN * kC * 4);   // (B,N,C)
  __bf16* WQb = (__bf16*)alloc((size_t)kCQ * kC * 2);
  __bf16* WKb = (__bf16*)alloc((size_t)kCQ * kC * 2);
  __bf16* WVb = (__bf16*)alloc((size_t)kC * kC * 2);

  cvt_weights<<<(kC * kC + 255) / 256, 256, 0, stream>>>(wq, wk, wv, WQb, WKb, WVb);
  pool_ln<<<kB * kHp * 8, 256, 0, stream>>>(x, lniw, lnib, XN);
  proj_qk<<<dim3(kN / 64, kB), 128, 0, stream>>>(XN, WQb, WKb, bq, bk, Qw, Kw);
  proj_v<<<dim3(kN / 64, kB * 8), 128, 0, stream>>>(XN, WVb, bv, Vw);
  flash_attn<<<dim3(kN / 16, kB), 128, 0, stream>>>(Qw, Kw, Vw, Ow);
  ln_out_up<<<kB * kHp, 256, 0, stream>>>(Ow, x, gamma, lnow, lnob, out);
}
